// CRF_31284541784224
// MI455X (gfx1250) — compile-verified, matrix operand order checked by reading, not run on hardware
//
#include <hip/hip_runtime.h>
#include <cstdint>
#include <cstddef>

// Problem constants (match reference)
#define B_   256
#define L_   1024
#define D_   126
#define T_   128        // D + 2
#define NEGV (-10000.0f)

#define LOG2E 1.4426950408889634f
#define LN2   0.6931471805599453f

typedef __attribute__((ext_vector_type(16))) _Float16 v16h;
typedef __attribute__((ext_vector_type(8)))  _Float16 v8h;
typedef __attribute__((ext_vector_type(8)))  float    v8f;

// ---- raw transcendental ops (v_exp_f32 / v_log_f32 are base-2) -------------
// Device pass: raw hardware builtins (no denorm/fixup code on the critical
// path). Host pass: portable HIP math so the host semantic check passes.
#if defined(__HIP_DEVICE_COMPILE__) && __has_builtin(__builtin_amdgcn_exp2f)
#define EXP2F(x) __builtin_amdgcn_exp2f(x)
#else
#define EXP2F(x) exp2f(x)
#endif
#if defined(__HIP_DEVICE_COMPILE__) && __has_builtin(__builtin_amdgcn_logf)
#define LOG2F(x) __builtin_amdgcn_logf(x)
#else
#define LOG2F(x) log2f(x)
#endif

// ---- CDNA5 async global->LDS copy path (ASYNCcnt) --------------------------
// Probe result (round 2 diagnostic): the b64 builtin takes
// (addrspace(1) int __vector(2)*, addrspace(3) int __vector(2)*, imm, imm).
#if defined(__HIP_DEVICE_COMPILE__) && \
    __has_builtin(__builtin_amdgcn_global_load_async_to_lds_b64) && \
    __has_builtin(__builtin_amdgcn_s_wait_asynccnt)
#define USE_ASYNC_LDS 1
typedef int v2i_pay __attribute__((__vector_size__(8)));
typedef __attribute__((address_space(1))) v2i_pay g_v2i;
typedef __attribute__((address_space(3))) v2i_pay l_v2i;
__device__ __forceinline__ void async_copy_b64(const float* g, float* l) {
  __builtin_amdgcn_global_load_async_to_lds_b64((g_v2i*)g, (l_v2i*)l, 0, 0);
}
__device__ __forceinline__ void async_wait0() {
  __builtin_amdgcn_s_wait_asynccnt(0);
}
#else
#define USE_ASYNC_LDS 0
#endif

// ---------------------------------------------------------------------------
// Kernel A: Et[i*T + j] = (f16) exp(transitions[i*T + j])
// ---------------------------------------------------------------------------
__global__ void crf_exp_trans(const float* __restrict__ trans,
                              _Float16* __restrict__ Et) {
  int idx = blockIdx.x * blockDim.x + threadIdx.x;
  for (int k = idx; k < T_ * T_; k += gridDim.x * blockDim.x)
    Et[k] = (_Float16)EXP2F(trans[k] * LOG2E);
}

// ---------------------------------------------------------------------------
// Kernel B: forward scan. One block = 16 batches, 128 threads (4 waves).
// alpha_new[b,i] = logit[b,i] + M_b + ln( sum_j e[b,j] * E[i,j] ),
// e[b,j] = exp(alpha[b,j] - M_b) -> (16x128)@(128x128) GEMM per step via
// v_wmma_f32_16x16x32_f16 with E^T fragments resident in VGPRs.
// Logits double-buffered in LDS, prefetched with async-to-LDS copies.
// ---------------------------------------------------------------------------
__global__ void __launch_bounds__(128)
crf_scan(const float* __restrict__ x,
         const float* __restrict__ trans,
         const float* __restrict__ xmask,
         const _Float16* __restrict__ Et,
         float* __restrict__ partition) {
  __shared__ float    s_alpha[16][T_];      // 8 KB
  __shared__ _Float16 s_e[16][T_];          // 4 KB
  __shared__ float    s_logit[2][16][T_];   // 16 KB (double buffer)
  __shared__ float    s_M[16];
  __shared__ float    s_mask[2][16];

  const int tid  = threadIdx.x;
  const int lane = tid & 31;
  const int w    = tid >> 5;             // wave 0..3 -> columns [32w, 32w+32)
  const int b0   = blockIdx.x * 16;

  const int bt = tid >> 3;               // batch within tile (8 threads/batch)
  const int j0 = (tid & 7) * 16;         // 16 tags per thread

  // alpha0: NEG except START (= T-2); constant NEG pads of logit buffers
  for (int e = 0; e < 16; ++e)
    s_alpha[bt][j0 + e] = ((j0 + e) == (T_ - 2)) ? 0.0f : NEGV;
  if (tid < 32) {
    const int bb = tid & 15, bf = tid >> 4;
    s_logit[bf][bb][126] = NEGV;
    s_logit[bf][bb][127] = NEGV;
  }

  // Resident B fragments: B[k=j, n=i] = E[i,j] = Et[i*T + j].
  v16h bfrag[2][4];
  {
    const int off = (lane < 16) ? 0 : 8;
    for (int t = 0; t < 2; ++t) {
      const int i = w * 32 + t * 16 + (lane & 15);
      for (int kc = 0; kc < 4; ++kc) {
        const _Float16* row = Et + (size_t)i * T_ + kc * 32 + off;
        v8h lo = *(const v8h*)(row);
        v8h hi = *(const v8h*)(row + 16);
        v16h f;
        for (int q = 0; q < 8; ++q) { f[q] = lo[q]; f[q + 8] = hi[q]; }
        bfrag[t][kc] = f;
      }
    }
  }

  // stage step-0 logits + mask into buffer 0
  {
    const size_t base0 = (size_t)(b0 + bt) * L_ * D_;
    for (int e = 0; e < 16; ++e) {
      const int d = j0 + e;
      if (d < D_) s_logit[0][bt][d] = x[base0 + d];
    }
    if (tid < 16) s_mask[0][tid] = xmask[(size_t)(b0 + tid) * L_];
  }
  __syncthreads();

  for (int l = 0; l < L_; ++l) {
    const int buf = l & 1;

    // ---- issue prefetch of step l+1 into the other buffer ----
    float mnext = 1.0f;
#if USE_ASYNC_LDS
    if (l + 1 < L_) {
      const float* gsrc = x + ((size_t)(b0 + bt) * L_ + (l + 1)) * D_ + j0;
      float*       ldst = &s_logit[buf ^ 1][bt][j0];
      for (int q = 0; q < 7; ++q)                 // 14 floats, 8B aligned
        async_copy_b64(gsrc + q * 2, ldst + q * 2);
      if (j0 < 112)                               // last pair (d=j0+14,15)
        async_copy_b64(gsrc + 14, ldst + 14);
      if (tid < 16) mnext = xmask[(size_t)(b0 + tid) * L_ + (l + 1)];
    }
#else
    float pre[16];
    if (l + 1 < L_) {
      const size_t nb = ((size_t)(b0 + bt) * L_ + (l + 1)) * D_;
      for (int e = 0; e < 16; ++e) {
        const int d = j0 + e;
        pre[e] = (d < D_) ? x[nb + d] : NEGV;
      }
      if (tid < 16) mnext = xmask[(size_t)(b0 + tid) * L_ + (l + 1)];
    }
#endif

    // ---- per-batch max over alpha (xor tree in an aligned 8-lane group) ----
    float mx = NEGV;
    for (int e = 0; e < 16; ++e) mx = fmaxf(mx, s_alpha[bt][j0 + e]);
    mx = fmaxf(mx, __shfl_xor(mx, 1, 8));
    mx = fmaxf(mx, __shfl_xor(mx, 2, 8));
    mx = fmaxf(mx, __shfl_xor(mx, 4, 8));
    if ((tid & 7) == 0) s_M[bt] = mx;            // for the cross-wave epilogue
    const float Mb = mx;                          // all 8 lanes hold the max
    for (int e = 0; e < 16; ++e)
      s_e[bt][j0 + e] = (_Float16)EXP2F((s_alpha[bt][j0 + e] - Mb) * LOG2E);
    __syncthreads();

    // ---- GEMM: acc[b,i] = sum_j e[b,j] * E[i,j]; 8 WMMAs per wave ----
    v8f acc0 = {}; v8f acc1 = {};
    {
      const int m   = lane & 15;
      const int off = (lane < 16) ? 0 : 8;
      for (int kc = 0; kc < 4; ++kc) {
        const _Float16* row = &s_e[m][kc * 32 + off];
        v8h lo = *(const v8h*)(row);
        v8h hi = *(const v8h*)(row + 16);
        v16h a;
        for (int q = 0; q < 8; ++q) { a[q] = lo[q]; a[q + 8] = hi[q]; }
        acc0 = __builtin_amdgcn_wmma_f32_16x16x32_f16(false, a, false, bfrag[0][kc],
                                                      (short)0, acc0, false, false);
        acc1 = __builtin_amdgcn_wmma_f32_16x16x32_f16(false, a, false, bfrag[1][kc],
                                                      (short)0, acc1, false, false);
      }
    }

    // ---- epilogue: alpha_new = logit + M + ln2*log2(acc); masked select ----
    {
      const int ln  = lane & 15;
      const int mhi = (lane >= 16) ? 8 : 0;
      for (int r = 0; r < 8; ++r) {
        const int   m   = r + mhi;
        const float Mm  = s_M[m];
        const float msk = s_mask[buf][m];
        const int   i0  = w * 32 + ln;
        const int   i1  = i0 + 16;
        const float n0  = s_logit[buf][m][i0] + Mm + LN2 * LOG2F(acc0[r]);
        const float n1  = s_logit[buf][m][i1] + Mm + LN2 * LOG2F(acc1[r]);
        s_alpha[m][i0] = msk * n0 + (1.0f - msk) * s_alpha[m][i0];
        s_alpha[m][i1] = msk * n1 + (1.0f - msk) * s_alpha[m][i1];
      }
    }

    // ---- commit prefetch for next step ----
#if USE_ASYNC_LDS
    if (tid < 16 && l + 1 < L_) s_mask[buf ^ 1][tid] = mnext;
    async_wait0();
#else
    if (l + 1 < L_) {
      for (int e = 0; e < 16; ++e) {
        const int d = j0 + e;
        if (d < D_) s_logit[buf ^ 1][bt][d] = pre[e];
      }
      if (tid < 16) s_mask[buf ^ 1][tid] = mnext;
    }
#endif
    __syncthreads();
  }

  // ---- partition[b] = LSE_i( alpha[b,i] + trans[STOP, i] ) ----
  {
    float v[16];
    float mx = NEGV;
    for (int e = 0; e < 16; ++e) {
      v[e] = s_alpha[bt][j0 + e] + trans[(T_ - 1) * T_ + j0 + e];
      mx = fmaxf(mx, v[e]);
    }
    mx = fmaxf(mx, __shfl_xor(mx, 1, 8));
    mx = fmaxf(mx, __shfl_xor(mx, 2, 8));
    mx = fmaxf(mx, __shfl_xor(mx, 4, 8));
    float s = 0.0f;
    for (int e = 0; e < 16; ++e) s += EXP2F((v[e] - mx) * LOG2E);
    s += __shfl_xor(s, 1, 8);
    s += __shfl_xor(s, 2, 8);
    s += __shfl_xor(s, 4, 8);
    if ((tid & 7) == 0) partition[b0 + bt] = mx + LN2 * LOG2F(s);
  }
}

// ---------------------------------------------------------------------------
// Kernel C: gold-path score (emission + transitions) and final output.
// ---------------------------------------------------------------------------
__global__ void __launch_bounds__(128)
crf_score(const float* __restrict__ x,
          const float* __restrict__ trans,
          const float* __restrict__ xmask,
          const int* __restrict__ xlen,
          const int* __restrict__ tag,
          const float* __restrict__ partition,
          float* __restrict__ out) {
  __shared__ float red[128];
  const int b   = blockIdx.x;
  const int tid = threadIdx.x;

  float local = 0.0f;
  for (int l = tid; l < L_; l += 128) {
    const float m = xmask[(size_t)b * L_ + l];
    const int   t = tag[(size_t)b * L_ + l];
    local += m * x[((size_t)b * L_ + l) * D_ + t];        // emission
    if (l == 0) {
      local += trans[t * T_ + (T_ - 2)];                  // START -> tag0
    } else {
      const int tp = tag[(size_t)b * L_ + l - 1];
      local += m * trans[t * T_ + tp];                    // tag_{l-1} -> tag_l
    }
  }
  red[tid] = local;
  __syncthreads();
  for (int s = 64; s > 0; s >>= 1) {
    if (tid < s) red[tid] += red[tid + s];
    __syncthreads();
  }
  if (tid == 0) {
    const int len = xlen[b];
    const int tl  = tag[(size_t)b * L_ + len - 1];
    out[b] = red[0] + trans[(T_ - 1) * T_ + tl] - partition[b];
  }
}

// ---------------------------------------------------------------------------
extern "C" void kernel_launch(void* const* d_in, const int* in_sizes, int n_in,
                              void* d_out, int out_size, void* d_ws, size_t ws_size,
                              hipStream_t stream) {
  const float* x     = (const float*)d_in[0];   // (B, L, D) f32
  const float* trans = (const float*)d_in[1];   // (T, T)    f32
  const float* xmask = (const float*)d_in[2];   // (B, L)    f32
  const int*   xlen  = (const int*)d_in[3];     // (B,)      i32
  const int*   tag   = (const int*)d_in[4];     // (B, L)    i32
  float* out = (float*)d_out;                   // (B,)      f32

  _Float16* Et        = (_Float16*)d_ws;
  float*    partition = (float*)((char*)d_ws + (size_t)T_ * T_ * sizeof(_Float16));

  crf_exp_trans<<<16, 256, 0, stream>>>(trans, Et);
  crf_scan<<<B_ / 16, 128, 0, stream>>>(x, trans, xmask, Et, partition);
  crf_score<<<B_, 128, 0, stream>>>(x, trans, xmask, xlen, tag, partition, out);
}
